// SelfMultiHeadAttention_50929722196368
// MI455X (gfx1250) — matrix operand order repaired and checked by reading, not compiled, not run on hardware
//
#include <hip/hip_runtime.h>

// Problem dims
#define Bn 16
#define Sn 512
#define En 768
#define Hn 12
#define Dn 64
#define SCALING 0.125f  // 1/sqrt(64)

typedef __attribute__((ext_vector_type(16))) _Float16 v16h;
typedef __attribute__((ext_vector_type(8)))  _Float16 v8h;
typedef __attribute__((ext_vector_type(8)))  float    v8f;
typedef __attribute__((ext_vector_type(4)))  float    v4f;

union AFrag { v16h v; v8h h[2]; };

// ---- WMMA fragment helpers (layouts per CDNA5 ISA 7.12.2, wave32) ----
// A 16x32 f16: lane L holds row M=L&15; halves g=L>>4:
//   elems 0..7  -> K = 8g..8g+7,  elems 8..15 -> K = 16+8g..16+8g+7
__device__ inline v16h load_frag_a_f16(const _Float16* base, int ld, int lane) {
  const int M = lane & 15, g = lane >> 4;
  const _Float16* row = base + (size_t)M * ld;
  AFrag a;
  a.h[0] = *(const v8h*)(row + 8 * g);
  a.h[1] = *(const v8h*)(row + 16 + 8 * g);
  return a.v;
}

// B 32x16 f16: lane L holds col N=L&15; elems 0..7 -> K = 8g..8g+7, elems 8..15 -> K = 16+8g..16+8g+7
// (source tile stored K-contiguous per N, i.e. "B-transposed" row-major)
__device__ inline v16h load_frag_b_f16(const _Float16* base, int ld, int lane) {
  const int N = lane & 15, g = lane >> 4;
  const _Float16* row = base + (size_t)N * ld + 8 * g;
  AFrag b;
  b.h[0] = *(const v8h*)(row);
  b.h[1] = *(const v8h*)(row + 16);
  return b.v;
}

// A fragment from a read-once fp32 stream: non-temporal loads (TH=NT keeps
// the 192MB L2 free for the reused weight/kv/attn operands) + f16 convert.
__device__ inline v16h load_frag_a_f32(const float* base, int ld, int lane) {
  const int M = lane & 15, g = lane >> 4;
  const float* row = base + (size_t)M * ld;
  v4f x0 = __builtin_nontemporal_load((const v4f*)(row + 8 * g));
  v4f x1 = __builtin_nontemporal_load((const v4f*)(row + 8 * g + 4));
  v4f x2 = __builtin_nontemporal_load((const v4f*)(row + 16 + 8 * g));
  v4f x3 = __builtin_nontemporal_load((const v4f*)(row + 16 + 8 * g + 4));
  v16h a;
  a[0]  = (_Float16)x0.x; a[1]  = (_Float16)x0.y; a[2]  = (_Float16)x0.z; a[3]  = (_Float16)x0.w;
  a[4]  = (_Float16)x1.x; a[5]  = (_Float16)x1.y; a[6]  = (_Float16)x1.z; a[7]  = (_Float16)x1.w;
  a[8]  = (_Float16)x2.x; a[9]  = (_Float16)x2.y; a[10] = (_Float16)x2.z; a[11] = (_Float16)x2.w;
  a[12] = (_Float16)x3.x; a[13] = (_Float16)x3.y; a[14] = (_Float16)x3.z; a[15] = (_Float16)x3.w;
  return a;
}

__device__ inline v8f wmma_f16(v16h a, v16h b, v8f c) {
  return __builtin_amdgcn_wmma_f32_16x16x32_f16(false, a, false, b, (short)0, c,
                                                false, false);
}

// =====================================================================
// Kernel 1: q/k/v = X @ Wq + bq   (q additionally scaled by 1/sqrt(D))
// X in {query,key,value} fp32 [B*S, E]; Wq fp32 [E,E]
// q,k -> f16 [B,H,S,D]; v -> f16 transposed [B,H,D,S]
// grid: (128, 12, 3) blocks of 128 threads (4 waves, 64x64 tile each)
// =====================================================================
__global__ __launch_bounds__(128) void proj_qkv_kernel(
    const float* __restrict__ query, const float* __restrict__ key,
    const float* __restrict__ value, const float* __restrict__ Wq,
    const float* __restrict__ bq, _Float16* __restrict__ qh,
    _Float16* __restrict__ kh, _Float16* __restrict__ vT) {
  __shared__ _Float16 wT[64 * 40];  // W^T chunk: [n 0..63][k 0..31], pad->40
  const int tid = threadIdx.x, lane = tid & 31, wv = tid >> 5;
  const int m0 = blockIdx.x * 64, n0 = blockIdx.y * 64, tensor = blockIdx.z;
  const float* X = (tensor == 0) ? query : ((tensor == 1) ? key : value);

  v8f acc[4] = {};
  for (int kk = 0; kk < En / 32; ++kk) {
    const int k0 = kk * 32;
    __syncthreads();
    for (int i = tid; i < 32 * 64; i += 128) {
      int k = i >> 6, n = i & 63;
      wT[n * 40 + k] = (_Float16)Wq[(size_t)(k0 + k) * En + n0 + n];
    }
    if (kk + 1 < En / 32)  // pull next L2-resident weight chunk toward WGP
      __builtin_prefetch(Wq + (size_t)(k0 + 32) * En + n0 + (tid & 63), 0, 1);
    __syncthreads();
    v16h a = load_frag_a_f32(X + (size_t)(m0 + wv * 16) * En + k0, En, lane);
#pragma unroll
    for (int j = 0; j < 4; ++j) {
      v16h b = load_frag_b_f16(wT + j * 16 * 40, 40, lane);
      acc[j] = wmma_f16(a, b, acc[j]);
    }
  }

  const int g = lane >> 4, N = lane & 15;
#pragma unroll
  for (int j = 0; j < 4; ++j) {
    const int n = n0 + j * 16 + N;
    const int h = n >> 6, d = n & 63;
    const float bias = bq[n];
#pragma unroll
    for (int r = 0; r < 8; ++r) {
      const int m = m0 + wv * 16 + r + 8 * g;
      const int bb = m >> 9, s = m & (Sn - 1);
      const float val = acc[j][r] + bias;
      if (tensor == 0)
        qh[(((size_t)bb * Hn + h) * Sn + s) * Dn + d] = (_Float16)(val * SCALING);
      else if (tensor == 1)
        kh[(((size_t)bb * Hn + h) * Sn + s) * Dn + d] = (_Float16)val;
      else
        vT[(((size_t)bb * Hn + h) * Dn + d) * Sn + s] = (_Float16)val;
    }
  }
}

// =====================================================================
// Kernel 2: w = q k^T (+mask,+bias) -> out_w ; attn = softmax(w) -> out_attn
// grid: (S/16, B*H) blocks of 128 threads; each block does 16 T-rows
// attn_bias is read-once (201MB) -> NT loads; out_w write-once -> NT stores
// =====================================================================
__global__ __launch_bounds__(128) void attn_scores_kernel(
    const _Float16* __restrict__ qh, const _Float16* __restrict__ kh,
    const unsigned char* __restrict__ mask, const float* __restrict__ bias,
    float* __restrict__ out_w, float* __restrict__ out_attn) {
  __shared__ float wrow[16 * 520];  // 16 rows x 512 cols, padded stride
  const int tid = threadIdx.x, lane = tid & 31, wv = tid >> 5;
  const int t0 = blockIdx.x * 16, bh = blockIdx.y, b = bh / Hn;
  const _Float16* qrow = qh + ((size_t)bh * Sn + t0) * Dn;
  const v16h a0 = load_frag_a_f16(qrow, Dn, lane);
  const v16h a1 = load_frag_a_f16(qrow + 32, Dn, lane);
  const int g = lane >> 4, N = lane & 15;

  for (int iter = 0; iter < 8; ++iter) {
    const int n0 = iter * 64 + wv * 16;
    const _Float16* krow = kh + ((size_t)bh * Sn + n0) * Dn;
    v16h b0 = load_frag_b_f16(krow, Dn, lane);
    v16h b1 = load_frag_b_f16(krow + 32, Dn, lane);
    v8f acc = {};
    acc = wmma_f16(a0, b0, acc);
    acc = wmma_f16(a1, b1, acc);
#pragma unroll
    for (int r = 0; r < 8; ++r) {
      const int t = t0 + r + 8 * g;
      const int s = n0 + N;
      float w = acc[r] +
                __builtin_nontemporal_load(&bias[((size_t)bh * Sn + t) * Sn + s]);
      if (mask[b * Sn + s]) w = -__builtin_inff();
      __builtin_nontemporal_store(w, &out_w[((size_t)bh * Sn + t) * Sn + s]);
      wrow[(r + 8 * g) * 520 + s] = w;
    }
  }
  __syncthreads();

  // softmax over 512 cols: wave wv handles rows wv, wv+4, wv+8, wv+12
  for (int rr = 0; rr < 4; ++rr) {
    const int row = wv + rr * 4;
    float vals[16];
    float vmax = -__builtin_inff();
#pragma unroll
    for (int j = 0; j < 16; ++j) {
      vals[j] = wrow[row * 520 + lane + 32 * j];
      vmax = fmaxf(vmax, vals[j]);
    }
#pragma unroll
    for (int off = 16; off >= 1; off >>= 1)
      vmax = fmaxf(vmax, __shfl_xor(vmax, off, 32));
    float sum = 0.f;
#pragma unroll
    for (int j = 0; j < 16; ++j) {
      vals[j] = __expf(vals[j] - vmax);
      sum += vals[j];
    }
#pragma unroll
    for (int off = 16; off >= 1; off >>= 1) sum += __shfl_xor(sum, off, 32);
    const float inv = 1.0f / sum;
#pragma unroll
    for (int j = 0; j < 16; ++j)
      out_attn[((size_t)bh * Sn + t0 + row) * Sn + lane + 32 * j] = vals[j] * inv;
  }
}

// =====================================================================
// Kernel 3: o_heads = attn @ V ; attn fp32 (from d_out), V^T f16 [B,H,D,S]
// grid: (S/64, B*H) blocks of 128; wave wv owns rows t0+wv*16, loops all
// 4 d-strips => attn (201MB) is read exactly ONCE; only the L2-resident
// V^T (12.6MB) is touched by all waves. 4 WMMAs per A-fragment load.
// =====================================================================
__global__ __launch_bounds__(128) void attn_v_kernel(
    const float* __restrict__ attn, const _Float16* __restrict__ vT,
    _Float16* __restrict__ oh) {
  const int tid = threadIdx.x, lane = tid & 31, wv = tid >> 5;
  const int t0 = blockIdx.x * 64 + wv * 16, bh = blockIdx.y;
  const int b = bh / Hn, h = bh % Hn;
  const float* arow = attn + ((size_t)bh * Sn + t0) * Sn;
  const _Float16* vbase = vT + (size_t)bh * Dn * Sn;
  v8f acc[4] = {};
  for (int kk = 0; kk < Sn / 32; ++kk) {
    v16h a = load_frag_a_f32(arow + kk * 32, Sn, lane);
#pragma unroll
    for (int j = 0; j < 4; ++j) {
      v16h bb =
          load_frag_b_f16(vbase + (size_t)(j * 16) * Sn + kk * 32, Sn, lane);
      acc[j] = wmma_f16(a, bb, acc[j]);
    }
  }
  const int g = lane >> 4, N = lane & 15;
#pragma unroll
  for (int j = 0; j < 4; ++j) {
#pragma unroll
    for (int r = 0; r < 8; ++r) {
      const int t = t0 + r + 8 * g;
      oh[((size_t)(b * Sn + t)) * En + h * Dn + j * 16 + N] =
          (_Float16)acc[j][r];
    }
  }
}

// =====================================================================
// Kernel 4: o = o_heads @ Wo + bo  -> fp32 d_out region 0
// grid: (128, 12) blocks of 128 threads (4 waves, 64x64 tile)
// =====================================================================
__global__ __launch_bounds__(128) void out_proj_kernel(
    const _Float16* __restrict__ oh, const float* __restrict__ Wo,
    const float* __restrict__ bo, float* __restrict__ out) {
  __shared__ _Float16 wT[64 * 40];
  const int tid = threadIdx.x, lane = tid & 31, wv = tid >> 5;
  const int m0 = blockIdx.x * 64, n0 = blockIdx.y * 64;
  v8f acc[4] = {};
  for (int kk = 0; kk < En / 32; ++kk) {
    const int k0 = kk * 32;
    __syncthreads();
    for (int i = tid; i < 32 * 64; i += 128) {
      int k = i >> 6, n = i & 63;
      wT[n * 40 + k] = (_Float16)Wo[(size_t)(k0 + k) * En + n0 + n];
    }
    if (kk + 1 < En / 32)
      __builtin_prefetch(Wo + (size_t)(k0 + 32) * En + n0 + (tid & 63), 0, 1);
    __syncthreads();
    v16h a = load_frag_a_f16(oh + (size_t)(m0 + wv * 16) * En + k0, En, lane);
#pragma unroll
    for (int j = 0; j < 4; ++j) {
      v16h bb = load_frag_b_f16(wT + j * 16 * 40, 40, lane);
      acc[j] = wmma_f16(a, bb, acc[j]);
    }
  }
  const int g = lane >> 4, N = lane & 15;
#pragma unroll
  for (int j = 0; j < 4; ++j) {
    const int n = n0 + j * 16 + N;
    const float bias = bo[n];
#pragma unroll
    for (int r = 0; r < 8; ++r) {
      const int m = m0 + wv * 16 + r + 8 * g;
      out[(size_t)m * En + n] = acc[j][r] + bias;
    }
  }
}

extern "C" void kernel_launch(void* const* d_in, const int* in_sizes, int n_in,
                              void* d_out, int out_size, void* d_ws,
                              size_t ws_size, hipStream_t stream) {
  (void)in_sizes; (void)n_in; (void)out_size; (void)ws_size;
  const float* query = (const float*)d_in[0];
  const float* key_in = (const float*)d_in[1];
  const float* value = (const float*)d_in[2];
  const unsigned char* mask = (const unsigned char*)d_in[3];  // jnp.bool_
  const float* attn_bias = (const float*)d_in[4];
  const float* Wq = (const float*)d_in[5];
  const float* bq = (const float*)d_in[6];
  const float* Wo = (const float*)d_in[7];
  const float* bo = (const float*)d_in[8];

  // Outputs: (o [B,S,E], w [B*H,S,S], attn [B*H,S,S]) concatenated fp32
  float* out_o = (float*)d_out;
  float* out_w = out_o + (size_t)Bn * Sn * En;
  float* out_attn = out_w + (size_t)Bn * Hn * Sn * Sn;

  // Workspace: f16 q,k [B,H,S,D], v^T [B,H,D,S], o_heads [B,S,E]  (~50 MB)
  _Float16* qh = (_Float16*)d_ws;
  _Float16* kh = qh + (size_t)Bn * Hn * Sn * Dn;
  _Float16* vT = kh + (size_t)Bn * Hn * Sn * Dn;
  _Float16* oh = vT + (size_t)Bn * Hn * Sn * Dn;

  proj_qkv_kernel<<<dim3(Bn * Sn / 64, En / 64, 3), 128, 0, stream>>>(
      query, key_in, value, Wq, bq, qh, kh, vT);
  attn_scores_kernel<<<dim3(Sn / 16, Bn * Hn), 128, 0, stream>>>(
      qh, kh, mask, attn_bias, out_w, out_attn);
  attn_v_kernel<<<dim3(Sn / 64, Bn * Hn), 128, 0, stream>>>(out_attn, vT, oh);
  out_proj_kernel<<<dim3(Bn * Sn / 64, En / 64), 128, 0, stream>>>(oh, Wo, bo,
                                                                   out_o);
}
